// DGCNNEncoder_10187662426540
// MI455X (gfx1250) — matrix operand order repaired and checked by reading, not compiled, not run on hardware
//
#include <hip/hip_runtime.h>
#include <hip/hip_bf16.h>

// ---------------------------------------------------------------------------
// ChebConv(K=3) x3 for gfx1250: SpMM with fp32 HW atomics (L2-resident gathers)
// + bf16 WMMA dense GEMMs (v_wmma_f32_16x16x32_bf16), f32 accumulate.
// ---------------------------------------------------------------------------

typedef __bf16 bf16_t;
typedef bf16_t v16bf __attribute__((ext_vector_type(16)));
typedef float  v8f   __attribute__((ext_vector_type(8)));
typedef unsigned int u32x4 __attribute__((ext_vector_type(4)));

#define NN 100000   // nodes
#define NE 1600000  // edges

__device__ __forceinline__ float atom_add(float* p, float v) {
  return unsafeAtomicAdd(p, v);  // global_atomic_add_f32 (no CAS loop)
}

// ---------------- graph normalization ----------------
__global__ void k_degree(const int* __restrict__ src, const int* __restrict__ dst,
                         float* __restrict__ deg, int E) {
  int i = blockIdx.x * blockDim.x + threadIdx.x;
  int stride = gridDim.x * blockDim.x;
  for (; i < E; i += stride) {
    int s = src[i];
    if (s != dst[i]) atom_add(&deg[s], 1.0f);
  }
}

__global__ void k_dinv(float* __restrict__ deg, int n) {
  int i = blockIdx.x * blockDim.x + threadIdx.x;
  if (i < n) {
    float d = deg[i];
    deg[i] = (d > 0.0f) ? rsqrtf(d) : 0.0f;
  }
}

__global__ void k_wedge(const int* __restrict__ src, const int* __restrict__ dst,
                        const float* __restrict__ dinv, float* __restrict__ w, int E) {
  int i = blockIdx.x * blockDim.x + threadIdx.x;
  int stride = gridDim.x * blockDim.x;
  for (; i < E; i += stride) {
    int s = src[i], d = dst[i];
    w[i] = (s != d) ? (-dinv[s] * dinv[d]) : 0.0f;
  }
}

// ---------------- SpMM: out[dst,:] += w[e] * h[src,:] (wave per edge) -------
__global__ void k_spmm(const int* __restrict__ src, const int* __restrict__ dst,
                       const float* __restrict__ w, const float* __restrict__ h,
                       float* __restrict__ out, int E, int C) {
  const int lane = threadIdx.x & 31;
  int wid = (blockIdx.x * blockDim.x + threadIdx.x) >> 5;
  int nw  = (gridDim.x * blockDim.x) >> 5;
  for (int e = wid; e < E; e += nw) {
    float we = w[e];
    if (we == 0.0f) continue;
    const float* hs = h + (size_t)src[e] * C;
    float* od = out + (size_t)dst[e] * C;
    for (int c = lane; c < C; c += 32)
      atom_add(&od[c], we * hs[c]);
  }
}

// ---------------- pack activations: [tx0 | tx1 | 2*tx2 - tx0] -> bf16 -------
__global__ void k_pack_a(const float* __restrict__ t0, const float* __restrict__ t1,
                         const float* __restrict__ t2, bf16_t* __restrict__ A,
                         int cshift) {
  const int C = 1 << cshift;
  size_t total = (size_t)NN << cshift;
  size_t i = (size_t)blockIdx.x * blockDim.x + threadIdx.x;
  size_t stride = (size_t)gridDim.x * blockDim.x;
  for (; i < total; i += stride) {
    size_t n = i >> cshift;
    int c = (int)(i & (C - 1));
    size_t base = n * (size_t)(3 * C);
    float a0 = t0[i];
    A[base + c]         = (bf16_t)a0;
    A[base + C + c]     = (bf16_t)t1[i];
    A[base + 2 * C + c] = (bf16_t)(2.0f * t2[i] - a0);
  }
}

// ---------------- pack weights: W[3,Cin,Cout] -> B^T [Cout, 3*Cin] bf16 -----
__global__ void k_pack_b(const float* __restrict__ W, bf16_t* __restrict__ Bt,
                         int Cin, int Cout) {
  int total = 3 * Cin * Cout;
  int i = blockIdx.x * blockDim.x + threadIdx.x;
  int stride = gridDim.x * blockDim.x;
  for (; i < total; i += stride) {
    int co = i % Cout;
    int kk = i / Cout;  // kk = kblock*Cin + ci ; W flat index == i
    Bt[(size_t)co * (3 * Cin) + kk] = (bf16_t)W[i];
  }
}

// ---------------- bf16 WMMA GEMM: out = relu?(A @ B^T + bias) ---------------
// A: [NN, K3] bf16 row-major; Bt: [COUT, K3] bf16 row-major (B transposed).
// One wave computes one 16x16 tile, K stepped by 32.
template <int K3, int COUT, bool RELU>
__global__ __launch_bounds__(256) void k_gemm(const bf16_t* __restrict__ A,
                                              const bf16_t* __restrict__ Bt,
                                              const float* __restrict__ bias,
                                              float* __restrict__ out) {
  constexpr int ROW_TILES = NN / 16;     // 6250 (exact)
  constexpr int COL_TILES = COUT / 16;
  const int lane = threadIdx.x & 31;
  const int wid  = threadIdx.x >> 5;
  const int t = blockIdx.x * 8 + wid;
  if (t >= ROW_TILES * COL_TILES) return;  // wave-uniform exit (EXEC stays full)

  const int row0 = (t / COL_TILES) * 16;
  const int n0   = (t % COL_TILES) * 16;
  const int hf = lane >> 4;   // lane half selects K sub-range per ISA layout
  const int ml = lane & 15;

  const bf16_t* arow = A  + (size_t)(row0 + ml) * K3;
  const bf16_t* brow = Bt + (size_t)(n0 + ml) * K3;

  v8f acc = {};
#pragma unroll
  for (int k = 0; k < K3; k += 32) {
    // A 16x32 bf16 fragment: lane half 0 -> K {0..7,16..23}, half 1 -> +8
    union { u32x4 u[2]; v16bf v; } fa, fb;
    fa.u[0] = *reinterpret_cast<const u32x4*>(arow + k + 8 * hf);
    fa.u[1] = *reinterpret_cast<const u32x4*>(arow + k + 16 + 8 * hf);
    // B 32x16 bf16 fragment: lane half 0 -> K {0..7,16..23}, half 1 -> +8
    fb.u[0] = *reinterpret_cast<const u32x4*>(brow + k + 8 * hf);
    fb.u[1] = *reinterpret_cast<const u32x4*>(brow + k + 16 + 8 * hf);
    acc = __builtin_amdgcn_wmma_f32_16x16x32_bf16(
        /*neg_a=*/false, fa.v, /*neg_b=*/false, fb.v,
        /*c_mod=*/(short)0, acc, /*reuse_a=*/false, /*reuse_b=*/false);
  }

  // C/D layout: VGPR r -> M = row0 + 8*hf + r, N = n0 + (lane & 15)
#pragma unroll
  for (int r = 0; r < 8; ++r) {
    int m = row0 + hf * 8 + r;
    int n = n0 + ml;
    float v = acc[r] + bias[n];
    if (RELU) v = v > 0.0f ? v : 0.0f;
    out[(size_t)m * COUT + n] = v;
  }
}

// ---------------------------------------------------------------------------
extern "C" void kernel_launch(void* const* d_in, const int* in_sizes, int n_in,
                              void* d_out, int out_size, void* d_ws, size_t ws_size,
                              hipStream_t stream) {
  const float* x  = (const float*)d_in[0];
  const int*   ei = (const int*)d_in[1];
  // d_in[2] = batch (unused)
  const float* W1 = (const float*)d_in[3];
  const float* b1 = (const float*)d_in[4];
  const float* W2 = (const float*)d_in[5];
  const float* b2 = (const float*)d_in[6];
  const float* W3 = (const float*)d_in[7];
  const float* b3 = (const float*)d_in[8];

  const int E = in_sizes[1] / 2;
  const int* src = ei;
  const int* dst = ei + E;

  // ---- workspace carving ----
  char* ws = (char*)d_ws;
  size_t off = 0;
  auto carve = [&](size_t bytes) -> void* {
    void* p = ws + off;
    off = (off + bytes + 255) & ~(size_t)255;
    return p;
  };
  float*  wedge = (float*)carve((size_t)E * 4);
  float*  dinv  = (float*)carve((size_t)NN * 4);       // deg then rsqrt in place
  float*  tx1   = (float*)carve((size_t)NN * 128 * 4);
  float*  tx2   = (float*)carve((size_t)NN * 128 * 4);
  float*  hA    = (float*)carve((size_t)NN * 64 * 4);  // layer-1 output
  float*  hB    = (float*)carve((size_t)NN * 128 * 4); // layer-2 output
  bf16_t* Abuf  = (bf16_t*)carve((size_t)NN * 384 * 2);
  bf16_t* Bbuf  = (bf16_t*)carve((size_t)256 * 384 * 2);
  (void)ws_size; (void)n_in; (void)out_size;

  // ---- edge normalization ----
  hipMemsetAsync(dinv, 0, (size_t)NN * 4, stream);
  k_degree<<<1024, 256, 0, stream>>>(src, dst, dinv, E);
  k_dinv<<<(NN + 255) / 256, 256, 0, stream>>>(dinv, NN);
  k_wedge<<<2048, 256, 0, stream>>>(src, dst, dinv, wedge, E);

  // ---- Layer 1: 128 -> 64, ReLU ----
  hipMemsetAsync(tx1, 0, (size_t)NN * 128 * 4, stream);
  hipMemsetAsync(tx2, 0, (size_t)NN * 128 * 4, stream);
  k_spmm<<<4096, 256, 0, stream>>>(src, dst, wedge, x,   tx1, E, 128);
  k_spmm<<<4096, 256, 0, stream>>>(src, dst, wedge, tx1, tx2, E, 128);
  k_pack_a<<<4096, 256, 0, stream>>>(x, tx1, tx2, Abuf, 7);
  k_pack_b<<<(3 * 128 * 64 + 255) / 256, 256, 0, stream>>>(W1, Bbuf, 128, 64);
  k_gemm<384, 64, true><<<(6250 * 4 + 7) / 8, 256, 0, stream>>>(Abuf, Bbuf, b1, hA);

  // ---- Layer 2: 64 -> 128, ReLU ----
  hipMemsetAsync(tx1, 0, (size_t)NN * 64 * 4, stream);
  hipMemsetAsync(tx2, 0, (size_t)NN * 64 * 4, stream);
  k_spmm<<<4096, 256, 0, stream>>>(src, dst, wedge, hA,  tx1, E, 64);
  k_spmm<<<4096, 256, 0, stream>>>(src, dst, wedge, tx1, tx2, E, 64);
  k_pack_a<<<4096, 256, 0, stream>>>(hA, tx1, tx2, Abuf, 6);
  k_pack_b<<<(3 * 64 * 128 + 255) / 256, 256, 0, stream>>>(W2, Bbuf, 64, 128);
  k_gemm<192, 128, true><<<(6250 * 8 + 7) / 8, 256, 0, stream>>>(Abuf, Bbuf, b2, hB);

  // ---- Layer 3: 128 -> 256, no activation, write d_out ----
  hipMemsetAsync(tx1, 0, (size_t)NN * 128 * 4, stream);
  hipMemsetAsync(tx2, 0, (size_t)NN * 128 * 4, stream);
  k_spmm<<<4096, 256, 0, stream>>>(src, dst, wedge, hB,  tx1, E, 128);
  k_spmm<<<4096, 256, 0, stream>>>(src, dst, wedge, tx1, tx2, E, 128);
  k_pack_a<<<4096, 256, 0, stream>>>(hB, tx1, tx2, Abuf, 7);
  k_pack_b<<<(3 * 128 * 256 + 255) / 256, 256, 0, stream>>>(W3, Bbuf, 128, 256);
  k_gemm<384, 256, false><<<(6250 * 16 + 7) / 8, 256, 0, stream>>>(Abuf, Bbuf, b3,
                                                                   (float*)d_out);
}
